// DummyNodes_8237747274338
// MI455X (gfx1250) — compile-verified, matrix operand order checked by reading, not run on hardware
//
#include <hip/hip_runtime.h>
#include <math.h>

typedef __attribute__((ext_vector_type(16))) _Float16 v16h;
typedef __attribute__((ext_vector_type(8)))  float    v8f;

#define NN 30000
#define BG 16
#define DF 32
#define KB 8
#define HH 8
#define SCALE 0.17677669529663687f   /* 1/sqrt(32) */
#define DIST 5.0f
#define EPSL 1e-6f

__device__ __forceinline__ unsigned f2ord(float f) {
  unsigned u = __float_as_uint(f);
  return (u & 0x80000000u) ? ~u : (u | 0x80000000u);
}
__device__ __forceinline__ float ord2f(unsigned o) {
  unsigned u = (o & 0x80000000u) ? (o & 0x7FFFFFFFu) : ~o;
  return __uint_as_float(u);
}
__device__ __forceinline__ float silu_f(float y) { return y / (1.0f + expf(-y)); }

/* ---------------- K0: init atomic targets (rerun every call) ---------------- */
__global__ void k_init(unsigned* zmin_b, unsigned* zmax_b, unsigned* m_b) {
  int t = threadIdx.x;
  if (t < BG) { zmin_b[t] = 0xFFFFFFFFu; zmax_b[t] = 0u; }
  if (t < 2 * BG) m_b[t] = 0u;
}

/* ---------------- K1: per-graph z min/max via ordered-uint atomics ---------- */
__global__ void k_bounds(const float* __restrict__ pos, const int* __restrict__ batch,
                         unsigned* zmin_b, unsigned* zmax_b) {
  int n = blockIdx.x * blockDim.x + threadIdx.x;
  if (n >= NN) return;
  int g = batch[n];
  unsigned u = f2ord(pos[n * 3 + 2]);
  atomicMin(&zmin_b[g], u);
  atomicMax(&zmax_b[g], u);
}

/* ---------------- K1b: graph start offsets via binary search ---------------- */
__global__ void k_offsets(const int* __restrict__ batch, int* offs) {
  int g = threadIdx.x;
  if (g > BG) return;
  int lo = 0, hi = NN;
  while (lo < hi) { int mid = (lo + hi) >> 1; if (batch[mid] < g) lo = mid + 1; else hi = mid; }
  offs[g] = lo;
}

/* ---------------- K2: tiny constant precompute ------------------------------ */
__global__ void k_precompute(const float* __restrict__ init_dummy,
                             const float* __restrict__ Wq_dummy,
                             const float* __restrict__ Wdot,
                             const float* __restrict__ W2kd,
                             const float* __restrict__ Wq_graph,
                             float* qW, float* u_kd, float* Wqd) {
  __shared__ float qd_l[DF], qW_l[DF];
  int t = threadIdx.x;
  if (t < DF) {
    float a = 0.f;
    for (int q = 0; q < DF; ++q) a += init_dummy[q] * Wq_dummy[q * DF + t];
    qd_l[t] = a;
  }
  __syncthreads();
  if (t < DF) {
    float a = 0.f;
    for (int q = 0; q < DF; ++q) a += qd_l[q] * Wdot[q * DF + t];
    qW_l[t] = a; qW[t] = a;
  }
  __syncthreads();
  { /* u_kd[h,d] = sum_e W2kd[h, d*32+e] * qW[e] */
    int h = t >> 5, d = t & 31;
    float a = 0.f;
    for (int e = 0; e < DF; ++e) a += W2kd[h * 1024 + d * DF + e] * qW_l[e];
    u_kd[t] = a;
  }
  for (int i = t; i < DF * DF; i += 256) { /* Wqd = Wq_graph @ Wdot */
    int r = i >> 5, c = i & 31;
    float a = 0.f;
    for (int q = 0; q < DF; ++q) a += Wq_graph[r * DF + q] * Wdot[q * DF + c];
    Wqd[i] = a;
  }
}

/* ---------------- K3: per-node scores + hvec_vd store + segment max --------- */
__global__ void k_scores(const float* __restrict__ pos, const int* __restrict__ batch,
                         const float* __restrict__ x, const float* __restrict__ bw,
                         const float* __restrict__ W1kd, const float* __restrict__ W1vd,
                         const float* __restrict__ u_kd,
                         const unsigned* __restrict__ zmin_b, const unsigned* __restrict__ zmax_b,
                         unsigned* m_b, float* s_e, float* hv_vd) {
  __shared__ unsigned mloc[2 * BG];
  int t = threadIdx.x;
  if (t < 2 * BG) mloc[t] = 0u;
  __syncthreads();
  int n = blockIdx.x * blockDim.x + t;
  if (n < NN) {
    int g = batch[n];
    float z = pos[n * 3 + 2];
    float len[2];
    len[0] = z - ord2f(zmin_b[g]) + DIST;
    len[1] = ord2f(zmax_b[g]) + DIST - z;
    float edge[2][KB];
    for (int s = 0; s < 2; ++s)
      for (int k = 0; k < KB; ++k)
        edge[s][k] = sinf(bw[k] * len[s]) / (len[s] + EPSL);
    float xr[DF];
    for (int d = 0; d < DF; ++d) xr[d] = x[n * DF + d];
    float th[HH];
    for (int h = 0; h < HH; ++h) {
      float a = 0.f;
      for (int d = 0; d < DF; ++d) a += xr[d] * u_kd[h * DF + d];
      th[h] = a;
    }
    for (int s = 0; s < 2; ++s) {
      float sv = 0.f;
      for (int h = 0; h < HH; ++h) {
        float ykd = 0.f, yvd = 0.f;
        for (int k = 0; k < KB; ++k) {
          ykd += edge[s][k] * W1kd[k * HH + h];
          yvd += edge[s][k] * W1vd[k * HH + h];
        }
        sv += silu_f(ykd) * th[h];
        hv_vd[(n * 2 + s) * HH + h] = silu_f(yvd);
      }
      sv *= SCALE;
      s_e[n * 2 + s] = sv;
      atomicMax(&mloc[g * 2 + s], f2ord(sv));
    }
  }
  __syncthreads();
  if (t < 2 * BG) atomicMax(&m_b[t], mloc[t]);
}

/* ---------------- K4: e = exp(s - m) (in place) ----------------------------- */
__global__ void k_exp(const int* __restrict__ batch, const unsigned* __restrict__ m_b,
                      float* s_e) {
  int n = blockIdx.x * blockDim.x + threadIdx.x;
  if (n >= NN) return;
  int g = batch[n];
  for (int s = 0; s < 2; ++s)
    s_e[n * 2 + s] = expf(s_e[n * 2 + s] - ord2f(m_b[g * 2 + s]));
}

/* ---------------- K5: A[g,s,h,d] deterministic segment sums ----------------- */
__global__ void k_accA(const float* __restrict__ e_buf, const float* __restrict__ hv_vd,
                       const float* __restrict__ x, const int* __restrict__ offs,
                       float* A_acc) {
  int t = blockIdx.x * blockDim.x + threadIdx.x;   /* 8192 threads */
  if (t >= BG * 2 * HH * DF) return;
  int g = t >> 9, s = (t >> 8) & 1, h = (t >> 5) & 7, d = t & 31;
  float acc = 0.f;
  int n0 = offs[g], n1 = offs[g + 1];
  for (int n = n0; n < n1; ++n)
    acc += e_buf[n * 2 + s] * hv_vd[(n * 2 + s) * HH + h] * x[n * DF + d];
  A_acc[t] = acc;
}

/* ---------------- K6: zsum, dummy_new, dM_kg, dM_vg (one block) ------------- */
__global__ void k_graphw(const float* __restrict__ e_buf, const int* __restrict__ offs,
                         const float* __restrict__ A_acc, const float* __restrict__ W2vd,
                         const float* __restrict__ init_dummy,
                         const float* __restrict__ W2kg, const float* __restrict__ W2vg,
                         float* dMkg, float* dMvg) {
  __shared__ float zs_l[2 * BG];
  __shared__ float dn_l[BG * 2 * DF];
  int t = threadIdx.x;   /* 1024 threads */
  if (t < 2 * BG) {
    int g = t >> 1, s = t & 1;
    float a = 0.f;
    for (int n = offs[g]; n < offs[g + 1]; ++n) a += e_buf[n * 2 + s];
    zs_l[t] = a;
  }
  __syncthreads();
  {
    int g = t >> 6, s = (t >> 5) & 1, eo = t & 31;
    const float* Ab = A_acc + (g * 2 + s) * (HH * DF);
    float acc = 0.f;
    for (int h = 0; h < HH; ++h)
      for (int d = 0; d < DF; ++d)
        acc += Ab[h * DF + d] * W2vd[h * 1024 + d * DF + eo];
    float zs = zs_l[g * 2 + s];
    dn_l[t] = init_dummy[eo] + ((zs > 0.f) ? acc / zs : 0.f);
  }
  __syncthreads();
  for (int i = t; i < BG * 2 * HH * DF; i += 1024) {
    int g = i >> 9, s = (i >> 8) & 1, h = (i >> 5) & 7, eo = i & 31;
    const float* dn = dn_l + (g * 2 + s) * DF;
    float a1 = 0.f, a2 = 0.f;
    for (int d = 0; d < DF; ++d) {
      float dv = dn[d];
      a1 += dv * W2kg[h * 1024 + d * DF + eo];
      a2 += dv * W2vg[h * 1024 + d * DF + eo];
    }
    dMkg[i] = a1; dMvg[i] = a2;
  }
}

/* ---------------- K7: qgW = x @ Wqd via WMMA, hi/lo f16 split --------------- */
__global__ void k_gemm_qgw(const float* __restrict__ x, const float* __restrict__ Wqd,
                           float* __restrict__ qgW) {
  const int lane = threadIdx.x & 31;
  const int wid  = (blockIdx.x * blockDim.x + threadIdx.x) >> 5;
  const int nw   = (gridDim.x * blockDim.x) >> 5;
  const int m    = lane & 15;
  const int ko   = (lane >> 4) << 3;   /* 0 or 8 */

  /* B tiles (two 16-col tiles of Wqd), hi/lo halves per CDNA5 16-bit B layout */
  v16h bh[2], bl[2];
#pragma unroll
  for (int tcol = 0; tcol < 2; ++tcol) {
#pragma unroll
    for (int i = 0; i < 8; ++i) {
      float f0 = Wqd[(ko + i) * DF + (16 * tcol + m)];
      float f1 = Wqd[(16 + ko + i) * DF + (16 * tcol + m)];
      _Float16 h0 = (_Float16)f0, h1 = (_Float16)f1;
      bh[tcol][i]     = h0; bl[tcol][i]     = (_Float16)(f0 - (float)h0);
      bh[tcol][8 + i] = h1; bl[tcol][8 + i] = (_Float16)(f1 - (float)h1);
    }
  }

  const int nTiles = NN / 16;   /* 1875 exactly */
  for (int tile = wid; tile < nTiles; tile += nw) {
    const float* row = x + (tile * 16 + m) * DF;
    v16h ah, al;
#pragma unroll
    for (int i = 0; i < 8; ++i) {
      float f0 = row[ko + i];
      float f1 = row[16 + ko + i];
      _Float16 h0 = (_Float16)f0, h1 = (_Float16)f1;
      ah[i]     = h0; al[i]     = (_Float16)(f0 - (float)h0);
      ah[8 + i] = h1; al[8 + i] = (_Float16)(f1 - (float)h1);
    }
#pragma unroll
    for (int tcol = 0; tcol < 2; ++tcol) {
      v8f c = {};
      c = __builtin_amdgcn_wmma_f32_16x16x32_f16(false, ah, false, bl[tcol], (short)0, c, false, false);
      c = __builtin_amdgcn_wmma_f32_16x16x32_f16(false, al, false, bh[tcol], (short)0, c, false, false);
      c = __builtin_amdgcn_wmma_f32_16x16x32_f16(false, ah, false, bh[tcol], (short)0, c, false, false);
      int mb = (lane < 16) ? 0 : 8;
#pragma unroll
      for (int j = 0; j < 8; ++j)
        qgW[(tile * 16 + mb + j) * DF + (16 * tcol + m)] = c[j];
    }
  }
}

/* ---------------- K8: phase-2 attention + output ---------------------------- */
__global__ void k_final(const float* __restrict__ pos, const int* __restrict__ batch,
                        const float* __restrict__ bw,
                        const float* __restrict__ W1kg, const float* __restrict__ W1vg,
                        const unsigned* __restrict__ zmin_b, const unsigned* __restrict__ zmax_b,
                        const float* __restrict__ x, const float* __restrict__ qgW,
                        const float* __restrict__ dMkg, const float* __restrict__ dMvg,
                        float* __restrict__ out) {
  int n = blockIdx.x * blockDim.x + threadIdx.x;
  if (n >= NN) return;
  int g = batch[n];
  float z = pos[n * 3 + 2];
  float len[2];
  len[0] = z - ord2f(zmin_b[g]) + DIST;
  len[1] = ord2f(zmax_b[g]) + DIST - z;
  float edge[2][KB];
  for (int s = 0; s < 2; ++s)
    for (int k = 0; k < KB; ++k)
      edge[s][k] = sinf(bw[k] * len[s]) / (len[s] + EPSL);
  float hvkg[2][HH], hvvg[2][HH];
  for (int s = 0; s < 2; ++s)
    for (int h = 0; h < HH; ++h) {
      float y1 = 0.f, y2 = 0.f;
      for (int k = 0; k < KB; ++k) {
        y1 += edge[s][k] * W1kg[k * HH + h];
        y2 += edge[s][k] * W1vg[k * HH + h];
      }
      hvkg[s][h] = silu_f(y1);
      hvvg[s][h] = silu_f(y2);
    }
  float q[DF];
  for (int e = 0; e < DF; ++e) q[e] = qgW[n * DF + e];
  float sg[2];
  for (int s = 0; s < 2; ++s) {
    float acc = 0.f;
    for (int h = 0; h < HH; ++h) {
      const float* dk = dMkg + ((g * 2 + s) * HH + h) * DF;
      float dot = 0.f;
      for (int e = 0; e < DF; ++e) dot += q[e] * dk[e];
      acc += hvkg[s][h] * dot;
    }
    sg[s] = acc * SCALE;
  }
  float mm = fmaxf(sg[0], sg[1]);
  float a0 = expf(sg[0] - mm), a1 = expf(sg[1] - mm);
  float inv = 1.0f / (a0 + a1);
  a0 *= inv; a1 *= inv;
  float w[2][HH];
  for (int h = 0; h < HH; ++h) { w[0][h] = a0 * hvvg[0][h]; w[1][h] = a1 * hvvg[1][h]; }
  for (int e = 0; e < DF; ++e) {
    float v = 0.f;
    for (int s = 0; s < 2; ++s)
      for (int h = 0; h < HH; ++h)
        v += w[s][h] * dMvg[((g * 2 + s) * HH + h) * DF + e];
    out[n * DF + e] = x[n * DF + e] + v;
  }
}

extern "C" void kernel_launch(void* const* d_in, const int* in_sizes, int n_in,
                              void* d_out, int out_size, void* d_ws, size_t ws_size,
                              hipStream_t stream) {
  const float* pos        = (const float*)d_in[0];
  const float* x          = (const float*)d_in[1];
  const int*   batch      = (const int*)  d_in[2];
  const float* bw         = (const float*)d_in[3];
  const float* init_dummy = (const float*)d_in[4];
  const float* Wq_dummy   = (const float*)d_in[5];
  const float* Wq_graph   = (const float*)d_in[6];
  const float* Wdot       = (const float*)d_in[7];
  const float* W1kd       = (const float*)d_in[8];
  const float* W2kd       = (const float*)d_in[9];
  const float* W1vd       = (const float*)d_in[10];
  const float* W2vd       = (const float*)d_in[11];
  const float* W1kg       = (const float*)d_in[12];
  const float* W2kg       = (const float*)d_in[13];
  const float* W1vg       = (const float*)d_in[14];
  const float* W2vg       = (const float*)d_in[15];
  float* out = (float*)d_out;

  float* ws = (float*)d_ws;
  unsigned* zmin_b = (unsigned*)(ws + 0);      /* 16  */
  unsigned* zmax_b = (unsigned*)(ws + 16);     /* 16  */
  unsigned* m_b    = (unsigned*)(ws + 32);     /* 32  */
  int*      offs   = (int*)     (ws + 64);     /* 17  (pad to 96) */
  float* qW    = ws + 96;                      /* 32   */
  float* u_kd  = ws + 128;                     /* 256  */
  float* Wqd   = ws + 384;                     /* 1024 */
  float* A_acc = ws + 1408;                    /* 8192 */
  float* dMkg  = ws + 9600;                    /* 8192 */
  float* dMvg  = ws + 17792;                   /* 8192 */
  float* s_e   = ws + 25984;                   /* N*2   = 60000  */
  float* hv_vd = ws + 85984;                   /* N*16  = 480000 */
  float* qgWb  = ws + 565984;                  /* N*32  = 960000 */

  const int TPB = 256;
  const int NB  = (NN + TPB - 1) / TPB;

  k_init<<<1, 64, 0, stream>>>(zmin_b, zmax_b, m_b);
  k_bounds<<<NB, TPB, 0, stream>>>(pos, batch, zmin_b, zmax_b);
  k_offsets<<<1, 32, 0, stream>>>(batch, offs);
  k_precompute<<<1, 256, 0, stream>>>(init_dummy, Wq_dummy, Wdot, W2kd, Wq_graph,
                                      qW, u_kd, Wqd);
  k_scores<<<NB, TPB, 0, stream>>>(pos, batch, x, bw, W1kd, W1vd, u_kd,
                                   zmin_b, zmax_b, m_b, s_e, hv_vd);
  k_exp<<<NB, TPB, 0, stream>>>(batch, m_b, s_e);
  k_accA<<<32, 256, 0, stream>>>(s_e, hv_vd, x, offs, A_acc);
  k_graphw<<<1, 1024, 0, stream>>>(s_e, offs, A_acc, W2vd, init_dummy, W2kg, W2vg,
                                   dMkg, dMvg);
  k_gemm_qgw<<<64, 256, 0, stream>>>(x, Wqd, qgWb);
  k_final<<<NB, TPB, 0, stream>>>(pos, batch, bw, W1kg, W1vg, zmin_b, zmax_b,
                                  x, qgWb, dMkg, dMvg, out);
}
